// TransformerEncoderLayerPreLN_2748779070108
// MI455X (gfx1250) — matrix operand order repaired and checked by reading, not compiled, not run on hardware
//
#include <hip/hip_runtime.h>

typedef _Float16 half_t;
typedef __attribute__((ext_vector_type(16))) _Float16 v16h;
typedef __attribute__((ext_vector_type(8)))  float    v8f;
typedef __attribute__((ext_vector_type(4)))  unsigned int u32x4;
typedef __attribute__((ext_vector_type(8)))  int i32x8;
typedef __attribute__((ext_vector_type(4)))  int i32x4;

#define WMMA_F32_F16(a, b, c) \
    __builtin_amdgcn_wmma_f32_16x16x32_f16(false, (a), false, (b), (short)0, (c), false, false)

// Async global->LDS copy, 16 bytes per lane (tracked by ASYNCcnt).
#define ASYNC_COPY_B128(ldsoff, gaddr)                                        \
    asm volatile("global_load_async_to_lds_b128 %0, %1, off"                  \
                 :: "v"(ldsoff), "v"(gaddr) : "memory")

__device__ inline void wait_async0() {
#if __has_builtin(__builtin_amdgcn_s_wait_asynccnt)
    __builtin_amdgcn_s_wait_asynccnt(0);
#else
    asm volatile("s_wait_asynccnt 0x0" ::: "memory");
#endif
}

__device__ inline void wait_tensor0() {
#if __has_builtin(__builtin_amdgcn_s_wait_tensorcnt)
    __builtin_amdgcn_s_wait_tensorcnt(0);
#else
    asm volatile("s_wait_tensorcnt 0x0" ::: "memory");
#endif
}

#if __has_builtin(__builtin_amdgcn_tensor_load_to_lds)
#define HAVE_TDM 1
#endif

__device__ inline unsigned lds_off_u32(const void* p) {
    return (unsigned)(size_t)p;  // low 32 bits of flat shared addr == LDS offset
}

// ---------------------------------------------------------------------------
// 16-lane xor-butterfly reductions. v_permlane16_b32 (VALU cross-lane, no
// LDS/dscnt stall) when available; each step's 4-bit lane selects encode
// lane^o within each row of 16 (both rows share the pattern).
// ---------------------------------------------------------------------------
#if __has_builtin(__builtin_amdgcn_permlane16)
template <unsigned LO, unsigned HI>
__device__ inline float permx16(float v) {
    int i = __float_as_int(v);
    i = __builtin_amdgcn_permlane16(i, i, (int)LO, (int)HI, false, false);
    return __int_as_float(i);
}
__device__ inline float rmax16(float v) {
    v = fmaxf(v, permx16<0x67452301u, 0xEFCDAB89u>(v));  // xor 1
    v = fmaxf(v, permx16<0x54761032u, 0xDCFE98BAu>(v));  // xor 2
    v = fmaxf(v, permx16<0x32107654u, 0xBA98FEDCu>(v));  // xor 4
    v = fmaxf(v, permx16<0xFEDCBA98u, 0x76543210u>(v));  // xor 8
    return v;
}
__device__ inline float rsum16(float v) {
    v += permx16<0x67452301u, 0xEFCDAB89u>(v);
    v += permx16<0x54761032u, 0xDCFE98BAu>(v);
    v += permx16<0x32107654u, 0xBA98FEDCu>(v);
    v += permx16<0xFEDCBA98u, 0x76543210u>(v);
    return v;
}
#else
__device__ inline float rmax16(float v) {
#pragma unroll
    for (int o = 1; o < 16; o <<= 1) v = fmaxf(v, __shfl_xor(v, o, 16));
    return v;
}
__device__ inline float rsum16(float v) {
#pragma unroll
    for (int o = 1; o < 16; o <<= 1) v += __shfl_xor(v, o, 16);
    return v;
}
#endif

// ---------------------------------------------------------------------------
// Fragment loaders (CDNA5 wave32 WMMA layouts, ISA 7.12.2)
// ---------------------------------------------------------------------------
// A: 16x32 f16, storage row-major [16][lda] halves.
__device__ inline v16h load_frag_a(const half_t* p, int lda, int lane) {
    int row = lane & 15;
    int grp = (lane >> 4) & 1;
    v16h a;
#pragma unroll
    for (int v = 0; v < 8; ++v) {
        int k = ((v < 4) ? (2 * v) : (16 + 2 * (v - 4))) + grp * 8;
        a[2 * v]     = p[row * lda + k];
        a[2 * v + 1] = p[row * lda + k + 1];
    }
    return a;
}

// B: 32x16 f16 (KxN), storage [n][ldb] halves (N-major, contiguous K).
__device__ inline v16h load_frag_b(const half_t* p, int ldb, int lane) {
    int n   = lane & 15;
    int grp = (lane >> 4) & 1;
    v16h b;
#pragma unroll
    for (int v = 0; v < 8; ++v) {
        int k = ((v < 4) ? (2 * v) : (16 + 2 * (v - 4))) + grp * 8;
        b[2 * v]     = p[n * ldb + k];
        b[2 * v + 1] = p[n * ldb + k + 1];
    }
    return b;
}

// ---------------------------------------------------------------------------
// fp32 -> f16 convert (weights)
// ---------------------------------------------------------------------------
__global__ void cvt_f32_f16_kernel(const float* __restrict__ in,
                                   half_t* __restrict__ out, int n) {
    int stride = gridDim.x * blockDim.x;
    for (int i = blockIdx.x * blockDim.x + threadIdx.x; i < n; i += stride)
        out[i] = (half_t)in[i];
}

// ---------------------------------------------------------------------------
// Masked norm: one 256-thread block per (b,s) row of D=512 features.
// ---------------------------------------------------------------------------
__global__ __launch_bounds__(256) void masked_norm_kernel(
    const float* __restrict__ x, const float* __restrict__ vm,
    half_t* __restrict__ out) {
    __shared__ float red[256];
    const int row = blockIdx.x;
    const int tid = threadIdx.x;
    const float* xr = x + (size_t)row * 512;
    const bool valid = vm[(size_t)row * 512] != 0.0f;

    float a = xr[tid];
    float b = xr[tid + 256];

    red[tid] = a + b;
    __syncthreads();
    for (int o = 128; o > 0; o >>= 1) {
        if (tid < o) red[tid] += red[tid + o];
        __syncthreads();
    }
    const float mean = red[0] * (1.0f / 512.0f);
    __syncthreads();

    float d0 = a - mean, d1 = b - mean;
    red[tid] = d0 * d0 + d1 * d1;
    __syncthreads();
    for (int o = 128; o > 0; o >>= 1) {
        if (tid < o) red[tid] += red[tid + o];
        __syncthreads();
    }
    const float var  = red[0] * (1.0f / 512.0f);
    const float rstd = rsqrtf(var + 1e-5f);

    half_t* orow = out + (size_t)row * 512;
    if (valid) {
        orow[tid]       = (half_t)(d0 * rstd);
        orow[tid + 256] = (half_t)(d1 * rstd);
    } else {
        orow[tid]       = (half_t)0.0f;
        orow[tid + 256] = (half_t)0.0f;
    }
}

// ---------------------------------------------------------------------------
// Generic GEMM:  Y[m,n] = sum_k A[m,k] * W[n,k]  (+ fused epilogue by MODE)
// Block = 256 threads = 8 waves, tile 128(M) x 64(N), K step 64.
// Double-buffered LDS staging via GLOBAL_LOAD_ASYNC_TO_LDS_B128 (ASYNCcnt):
// next K tile's DMA is issued before computing the current one.
// MODE 0: QKV scatter to q(pre-scaled)/k/v [B,H,S,64] f16
// MODE 1: out-proj: src2 = src + nan_to_num(acc+bias)*mask       (fp32)
// MODE 2: FFN1: h = relu(acc+bias)                               (f16)
// MODE 3: FFN2: out = src2 + acc + bias                          (fp32)
// ---------------------------------------------------------------------------
template <int MODE>
__global__ __launch_bounds__(256) void gemm_kernel(
    const half_t* __restrict__ A, const half_t* __restrict__ W,
    const float* __restrict__ bias, int M, int N, int K,
    const float* __restrict__ addf, const float* __restrict__ vmask,
    float* __restrict__ outf,
    half_t* __restrict__ outh0, half_t* __restrict__ outh1,
    half_t* __restrict__ outh2) {
    __shared__ __align__(16) half_t As[2][128 * 64];
    __shared__ __align__(16) half_t Bs[2][64 * 64];

    const int tid  = threadIdx.x;
    const int wave = tid >> 5;
    const int lane = tid & 31;
    const int wm   = wave & 3;   // 4 waves along M
    const int wn   = wave >> 2;  // 2 waves along N
    const int m0   = blockIdx.y * 128;
    const int n0   = blockIdx.x * 64;
    const unsigned asBase = lds_off_u32(&As[0][0]);
    const unsigned bsBase = lds_off_u32(&Bs[0][0]);

    auto stage = [&](int buf, int k0) {
        const unsigned aB = asBase + (unsigned)buf * (128u * 64u * 2u);
        const unsigned bB = bsBase + (unsigned)buf * (64u * 64u * 2u);
        // A tile: 128 rows x 64 halves = 1024 x 16B chunks, 4 per thread
#pragma unroll
        for (int i = 0; i < 4; ++i) {
            int idx = tid + i * 256;
            int row = idx >> 3, c8 = idx & 7;
            const half_t* gp = A + (size_t)(m0 + row) * K + k0 + c8 * 8;
            ASYNC_COPY_B128(aB + (unsigned)idx * 16u,
                            (unsigned long long)(size_t)gp);
        }
        // B tile: 64 rows x 64 halves = 512 chunks, 2 per thread
#pragma unroll
        for (int i = 0; i < 2; ++i) {
            int idx = tid + i * 256;
            int row = idx >> 3, c8 = idx & 7;
            const half_t* gp = W + (size_t)(n0 + row) * K + k0 + c8 * 8;
            ASYNC_COPY_B128(bB + (unsigned)idx * 16u,
                            (unsigned long long)(size_t)gp);
        }
    };

    v8f c[2][2] = {};

    stage(0, 0);
    wait_async0();
    __syncthreads();

    int cur = 0;
    for (int k0 = 0; k0 < K; k0 += 64) {
        const int nxt = cur ^ 1;
        if (k0 + 64 < K) stage(nxt, k0 + 64);  // overlap DMA with WMMA

        v16h af[2][2], bf[2][2];
#pragma unroll
        for (int i = 0; i < 2; ++i)
#pragma unroll
            for (int kk = 0; kk < 2; ++kk)
                af[i][kk] = load_frag_a(&As[cur][(wm * 32 + i * 16) * 64 + kk * 32], 64, lane);
#pragma unroll
        for (int j = 0; j < 2; ++j)
#pragma unroll
            for (int kk = 0; kk < 2; ++kk)
                bf[j][kk] = load_frag_b(&Bs[cur][(wn * 32 + j * 16) * 64 + kk * 32], 64, lane);
#pragma unroll
        for (int i = 0; i < 2; ++i)
#pragma unroll
            for (int j = 0; j < 2; ++j) {
                c[i][j] = WMMA_F32_F16(af[i][0], bf[j][0], c[i][j]);
                c[i][j] = WMMA_F32_F16(af[i][1], bf[j][1], c[i][j]);
            }

        wait_async0();
        __syncthreads();
        cur = nxt;
    }

    // Epilogue
    const int grp = lane >> 4;
    const int nn  = lane & 15;
#pragma unroll
    for (int i = 0; i < 2; ++i) {
#pragma unroll
        for (int j = 0; j < 2; ++j) {
#pragma unroll
            for (int r = 0; r < 8; ++r) {
                const int m = m0 + wm * 32 + i * 16 + r + 8 * grp;
                const int n = n0 + wn * 32 + j * 16 + nn;
                float v = c[i][j][r] + bias[n];
                if constexpr (MODE == 0) {
                    const int sel = n >> 9;      // 0:q 1:k 2:v
                    const int d   = n & 511;
                    const int hh  = d >> 6;
                    const int hd  = d & 63;
                    const int bb  = m >> 10;
                    const int ss  = m & 1023;
                    if (sel == 0) v *= 0.125f;   // 1/sqrt(HD=64)
                    half_t* dst = (sel == 0) ? outh0 : (sel == 1) ? outh1 : outh2;
                    dst[((size_t)((bb * 8 + hh) * 1024 + ss) << 6) + hd] = (half_t)v;
                } else if constexpr (MODE == 1) {
                    if (v != v) v = 0.0f;        // nan_to_num
                    v = fminf(fmaxf(v, -1e30f), 1e30f);
                    const size_t o = (size_t)m * N + n;
                    outf[o] = addf[o] + v * vmask[o];
                } else if constexpr (MODE == 2) {
                    outh0[(size_t)m * N + n] = (half_t)fmaxf(v, 0.0f);
                } else {  // MODE == 3
                    const size_t o = (size_t)m * N + n;
                    outf[o] = addf[o] + v;
                }
            }
        }
    }
}

// ---------------------------------------------------------------------------
// Flash-style attention. Block = 128 threads = 4 waves; each wave owns 16
// query rows; block owns 64 query rows of one (b,h). Double-buffered tiles:
// K tile staged by the Tensor Data Mover (TENSORcnt) when available (else
// async-to-LDS), V tile staged transposed ([hd][key]); the next tile's TDM
// and V global loads are issued before the current tile's compute.
// q is pre-scaled by 1/sqrt(64); bias(b,h) uses batch index (b*H+h)%B.
// ---------------------------------------------------------------------------
__global__ __launch_bounds__(128) void attn_kernel(
    const half_t* __restrict__ q, const half_t* __restrict__ k,
    const half_t* __restrict__ v, const float* __restrict__ vmask,
    half_t* __restrict__ ctx) {
    __shared__ __align__(16) half_t Ks[2][64 * 64];    // [key][hd]
    __shared__ __align__(16) half_t VsT[2][64 * 64];   // [hd][key]
    __shared__ __align__(16) half_t Ps[4][16 * 64];
    __shared__ unsigned char padf[1024];

    const int b    = blockIdx.z;
    const int h    = blockIdx.y;
    const int qb   = blockIdx.x * 64;
    const int tid  = threadIdx.x;
    const int wave = tid >> 5;
    const int lane = tid & 31;
    const int grp  = lane >> 4;
    const int nn   = lane & 15;
    const int b2   = (b * 8 + h) & 7;  // (b*H + h) % B, B==H==8
    const size_t bh = (size_t)(b * 8 + h) * 1024;
    const unsigned ksLds = lds_off_u32(&Ks[0][0]);

    auto issueK = [&](int buf, int kb) {
#ifdef HAVE_TDM
        if (tid < 32) {  // wave 0 drives the TDM for the K tile (64x64 f16)
            unsigned long long ga =
                (unsigned long long)(size_t)(k + (bh + kb) * 64);
            u32x4 g0;
            g0.x = 1u;                                   // count=1, load mode
            g0.y = ksLds + (unsigned)buf * 8192u;        // lds_addr
            g0.z = (unsigned)(ga & 0xffffffffu);         // global_addr[31:0]
            g0.w = (unsigned)((ga >> 32) & 0x01ffffffu)  // global_addr[56:32]
                   | 0x80000000u;                        // type=2 (image)
            i32x8 g1;
            g1[0] = 0x00010000;          // data_size=2B, no multicast
            g1[1] = (int)(64u << 16);    // tensor_dim0 = 64  (bits 79:48)
            g1[2] = (int)(1024u << 16);  // tensor_dim1 = 1024 (bits 111:80)
            g1[3] = (int)(64u << 16);    // tile_dim0 = 64 (bits 127:112)
            g1[4] = 64;                  // tile_dim1 = 64 (bits 143:128)
            g1[5] = 64;                  // tensor_dim0_stride = 64 (bits 207:160)
            g1[6] = 0;
            g1[7] = 0;
            i32x4 gz = {0, 0, 0, 0};
#if __clang_major__ >= 23
            i32x8 gz8 = {};
            __builtin_amdgcn_tensor_load_to_lds(g0, g1, gz, gz, gz8, 0);
#else
            __builtin_amdgcn_tensor_load_to_lds(g0, g1, gz, gz, 0);
#endif
        }
#else
        const unsigned kB = ksLds + (unsigned)buf * 8192u;
#pragma unroll
        for (int i = 0; i < 4; ++i) {
            int idx = tid + i * 128;
            int key = idx >> 3, part = idx & 7;
            const half_t* gp = k + (bh + kb + key) * 64 + part * 8;
            ASYNC_COPY_B128(kB + (unsigned)idx * 16u,
                            (unsigned long long)(size_t)gp);
        }
#endif
    };
    auto waitK = [&]() {
#ifdef HAVE_TDM
        if (tid < 32) wait_tensor0();
#else
        wait_async0();
#endif
    };
    auto loadV = [&](int kb, int4* t) {
#pragma unroll
        for (int i = 0; i < 4; ++i) {
            int idx = tid + i * 128;
            int key = idx >> 3, part = idx & 7;
            t[i] = reinterpret_cast<const int4*>(v + (bh + kb + key) * 64)[part];
        }
    };
    auto storeV = [&](int buf, const int4* t) {
#pragma unroll
        for (int i = 0; i < 4; ++i) {
            int idx = tid + i * 128;
            int key = idx >> 3, part = idx & 7;
            const half_t* hp = reinterpret_cast<const half_t*>(&t[i]);
#pragma unroll
            for (int e = 0; e < 8; ++e)
                VsT[buf][(part * 8 + e) * 64 + key] = hp[e];
        }
    };

    // stage pad flags for batch b2 from value_mask (pad == mask 0)
    for (int j = tid; j < 1024; j += 128)
        padf[j] = (vmask[((size_t)b2 * 1024 + j) * 512] == 0.0f) ? 1 : 0;

    // query fragments (A layout) straight from global
    const int qrow = qb + wave * 16 + nn;
    const half_t* qp = q + (bh + qrow) * 64;
    v16h qa0, qa1;
#pragma unroll
    for (int vv = 0; vv < 8; ++vv) {
        int kk = ((vv < 4) ? (2 * vv) : (16 + 2 * (vv - 4))) + grp * 8;
        qa0[2 * vv]     = qp[kk];
        qa0[2 * vv + 1] = qp[kk + 1];
        qa1[2 * vv]     = qp[32 + kk];
        qa1[2 * vv + 1] = qp[32 + kk + 1];
    }

    // prologue: stage tile 0 into buffer 0
    {
        int4 tv[4];
        issueK(0, 0);
        loadV(0, tv);
        storeV(0, tv);
        waitK();
    }
    __syncthreads();

    bool pq[8];
#pragma unroll
    for (int r = 0; r < 8; ++r)
        pq[r] = padf[qb + wave * 16 + r + 8 * grp] != 0;

    float m_i[8], l_i[8];
#pragma unroll
    for (int r = 0; r < 8; ++r) { m_i[r] = -1e30f; l_i[r] = 0.0f; }
    v8f acc[4] = {};

    int cur = 0;
    for (int kb = 0; kb < 1024; kb += 64) {
        const int nxt  = cur ^ 1;
        const bool more = (kb + 64) < 1024;
        int4 tv[4];
        if (more) {
            issueK(nxt, kb + 64);   // TDM overlaps with compute below
            loadV(kb + 64, tv);     // global loads overlap with compute
        }

        // scores: 4 tiles of 16x16, K-dim = 64 (two WMMAs each)
        v8f st[4] = {};
#pragma unroll
        for (int t = 0; t < 4; ++t) {
            v16h kb0 = load_frag_b(&Ks[cur][(t * 16) * 64], 64, lane);
            v16h kb1 = load_frag_b(&Ks[cur][(t * 16) * 64 + 32], 64, lane);
            st[t] = WMMA_F32_F16(qa0, kb0, st[t]);
            st[t] = WMMA_F32_F16(qa1, kb1, st[t]);
        }

        // bias + tile row max
        float tmax[8];
#pragma unroll
        for (int r = 0; r < 8; ++r) tmax[r] = -1e30f;
#pragma unroll
        for (int t = 0; t < 4; ++t) {
            const bool pk = padf[kb + t * 16 + nn] != 0;
#pragma unroll
            for (int r = 0; r < 8; ++r) {
                float s = st[t][r];
                if (pq[r] && pk) s -= 1e9f;
                st[t][r] = s;
                tmax[r]  = fmaxf(tmax[r], s);
            }
        }
#pragma unroll
        for (int r = 0; r < 8; ++r) tmax[r] = rmax16(tmax[r]);

        // online softmax update
        float rs[8];
#pragma unroll
        for (int r = 0; r < 8; ++r) {
            const float nm = fmaxf(m_i[r], tmax[r]);
            const float sc = __expf(m_i[r] - nm);
            m_i[r] = nm;
            l_i[r] *= sc;
            rs[r]  = 0.0f;
#pragma unroll
            for (int j = 0; j < 4; ++j) acc[j][r] *= sc;
        }
#pragma unroll
        for (int t = 0; t < 4; ++t)
#pragma unroll
            for (int r = 0; r < 8; ++r) {
                const float p = __expf(st[t][r] - m_i[r]);
                st[t][r] = p;
                rs[r] += p;
            }
#pragma unroll
        for (int r = 0; r < 8; ++r) l_i[r] += rsum16(rs[r]);

        // reshape P (C layout -> A layout) through per-wave LDS slice
        half_t* ps = Ps[wave];
#pragma unroll
        for (int t = 0; t < 4; ++t)
#pragma unroll
            for (int r = 0; r < 8; ++r)
                ps[(r + 8 * grp) * 64 + t * 16 + nn] = (half_t)st[t][r];

        v16h pa0 = load_frag_a(ps, 64, lane);
        v16h pa1 = load_frag_a(ps + 32, 64, lane);
#pragma unroll
        for (int j = 0; j < 4; ++j) {
            v16h vb0 = load_frag_b(&VsT[cur][(j * 16) * 64], 64, lane);
            v16h vb1 = load_frag_b(&VsT[cur][(j * 16) * 64 + 32], 64, lane);
            acc[j] = WMMA_F32_F16(pa0, vb0, acc[j]);
            acc[j] = WMMA_F32_F16(pa1, vb1, acc[j]);
        }

        if (more) {
            storeV(nxt, tv);  // flush prefetched V into next buffer
            waitK();          // TDM / async K tile completion
        }
        __syncthreads();
        cur = nxt;
    }

    // normalize and scatter ctx into [B,S,D] f16 (D col = h*64 + hd)
#pragma unroll
    for (int r = 0; r < 8; ++r) {
        const float inv = 1.0f / l_i[r];
        const int row = qb + wave * 16 + r + 8 * grp;
#pragma unroll
        for (int j = 0; j < 4; ++j) {
            const int col = h * 64 + j * 16 + nn;
            ctx[((size_t)b * 1024 + row) * 512 + col] = (half_t)(acc[j][r] * inv);
        }
    }
}

// ---------------------------------------------------------------------------
// Host launcher
// ---------------------------------------------------------------------------
extern "C" void kernel_launch(void* const* d_in, const int* in_sizes, int n_in,
                              void* d_out, int out_size, void* d_ws, size_t ws_size,
                              hipStream_t stream) {
    (void)in_sizes; (void)n_in; (void)out_size; (void)ws_size;
    constexpr int B = 8, S = 1024, D = 512, H = 8, F = 2048;
    constexpr int M = B * S;  // 8192

    const float* src        = (const float*)d_in[0];
    const float* value_mask = (const float*)d_in[1];
    // d_in[2] = src_mask (bool) unused; d_in[3] = id_batch unused
    const float* in_proj_w  = (const float*)d_in[4];
    const float* in_proj_b  = (const float*)d_in[5];
    const float* out_w      = (const float*)d_in[6];
    const float* out_b      = (const float*)d_in[7];
    const float* w1         = (const float*)d_in[8];
    const float* b1         = (const float*)d_in[9];
    const float* w2         = (const float*)d_in[10];
    const float* b2         = (const float*)d_in[11];
    float* out = (float*)d_out;

    char* ws = (char*)d_ws;
    size_t off = 0;
    auto take = [&](size_t bytes) {
        void* p = ws + off;
        off = (off + bytes + 255) & ~(size_t)255;
        return p;
    };

    half_t* wqkv_h = (half_t*)take((size_t)3 * D * D * 2);
    half_t* wout_h = (half_t*)take((size_t)D * D * 2);
    half_t* w1_h   = (half_t*)take((size_t)F * D * 2);
    half_t* w2_h   = (half_t*)take((size_t)D * F * 2);
    half_t* xn_h   = (half_t*)take((size_t)M * D * 2);
    half_t* q_h    = (half_t*)take((size_t)M * D * 2);
    half_t* k_h    = (half_t*)take((size_t)M * D * 2);
    half_t* v_h    = (half_t*)take((size_t)M * D * 2);
    half_t* ctx_h  = (half_t*)take((size_t)M * D * 2);
    float*  src2   = (float*)take((size_t)M * D * 4);
    half_t* h1_h   = (half_t*)take((size_t)M * F * 2);

    // 1) weights -> f16
    cvt_f32_f16_kernel<<<512, 256, 0, stream>>>(in_proj_w, wqkv_h, 3 * D * D);
    cvt_f32_f16_kernel<<<512, 256, 0, stream>>>(out_w, wout_h, D * D);
    cvt_f32_f16_kernel<<<512, 256, 0, stream>>>(w1, w1_h, F * D);
    cvt_f32_f16_kernel<<<512, 256, 0, stream>>>(w2, w2_h, D * F);

    // 2) masked norm of src
    masked_norm_kernel<<<M, 256, 0, stream>>>(src, value_mask, xn_h);

    // 3) QKV projection (M=8192, N=1536, K=512) with q/k/v scatter
    gemm_kernel<0><<<dim3((3 * D) / 64, M / 128), 256, 0, stream>>>(
        xn_h, wqkv_h, in_proj_b, M, 3 * D, D,
        nullptr, nullptr, nullptr, q_h, k_h, v_h);

    // 4) attention
    attn_kernel<<<dim3(S / 64, H, B), 128, 0, stream>>>(q_h, k_h, v_h,
                                                        value_mask, ctx_h);

    // 5) out projection + nan_to_num*mask + residual -> src2
    gemm_kernel<1><<<dim3(D / 64, M / 128), 256, 0, stream>>>(
        ctx_h, wout_h, out_b, M, D, D,
        src, value_mask, src2, nullptr, nullptr, nullptr);

    // 6) masked norm of src2 (reuse xn_h)
    masked_norm_kernel<<<M, 256, 0, stream>>>(src2, value_mask, xn_h);

    // 7) FFN1: relu(xn2 @ w1.T + b1) -> f16
    gemm_kernel<2><<<dim3(F / 64, M / 128), 256, 0, stream>>>(
        xn_h, w1_h, b1, M, F, D,
        nullptr, nullptr, nullptr, h1_h, nullptr, nullptr);

    // 8) FFN2: out = src2 + h1 @ w2.T + b2
    gemm_kernel<3><<<dim3(D / 64, M / 128), 256, 0, stream>>>(
        h1_h, w2_h, b2, M, D, F,
        src2, nullptr, out, nullptr, nullptr, nullptr);
}
